// TransformerScratch_34978213658636
// MI455X (gfx1250) — matrix operand order repaired and checked by reading, not compiled, not run on hardware
//
#include <hip/hip_runtime.h>
#include <math.h>

// ---------------- problem constants ----------------
#define Dm    1024
#define DFFm  4096
#define Bb    32
#define Ss    96        // S == T == 96
#define Hh    16
#define DHh   64
#define NLAY  6
#define MT    (Bb * Ss) // 3072 rows for both streams
#define VT    32000

typedef __attribute__((ext_vector_type(16))) __bf16 v16bf;
typedef __attribute__((ext_vector_type(8)))  float  v8f;

union Frag { v16bf v; unsigned u[8]; };

// fp32 -> bf16 round-to-nearest-even, packed pair
__device__ __forceinline__ unsigned short f2bf(float f) {
  unsigned u = __float_as_uint(f);
  unsigned r = 0x7FFFu + ((u >> 16) & 1u);
  return (unsigned short)((u + r) >> 16);
}
__device__ __forceinline__ unsigned pack2(float a, float b) {
  return (unsigned)f2bf(a) | ((unsigned)f2bf(b) << 16);
}

// batch offset modes: 0 = linear z*stride, 1 = per-(b,h) head view into [B*S, D]
__device__ __forceinline__ long long batchOff(int mode, long long stride, int z) {
  if (mode == 1)
    return (long long)(z / Hh) * (long long)Ss * Dm + (long long)(z % Hh) * DHh;
  return stride * (long long)z;
}

// async copy of 16 bytes global -> LDS (ASYNCcnt-tracked, CDNA5)
__device__ __forceinline__ void async_g2l_b128(unsigned lds_addr, const float* gp) {
  asm volatile("global_load_async_to_lds_b128 %0, %1, off"
               :: "v"(lds_addr), "v"(gp) : "memory");
}
__device__ __forceinline__ void wait_async0() {
  asm volatile("s_wait_asynccnt 0" ::: "memory");
}

// ---------------------------------------------------------------------------
// Generic batched GEMM: C = act(A[M,K] x B + bias), bf16 WMMA, fp32 accumulate
// transB==0: B is [K,N] row-major; transB==1: B is [N,K] row-major (for Q.K^T)
// Tile 64x64x32, 256 threads = 8 waves; wave (wm 0..3, wn 0..1) owns a 16x32
// output slab = two v_wmma_f32_16x16x32_bf16 accumulators sharing the A frag.
// Full interior tiles take an async-to-LDS (ASYNCcnt) staging path with no
// bounds checks; edge tiles (the 96-wide attention GEMMs) take a guarded path.
// ---------------------------------------------------------------------------
__global__ void __launch_bounds__(256)
gemm_bf16_wmma(const float* __restrict__ A, const float* __restrict__ Bm,
               float* __restrict__ C, const float* __restrict__ bias,
               int M, int N, int K, int lda, int ldb, int ldc,
               int transB, int relu,
               long long batA, long long batB, long long batC,
               int modeA, int modeB, int modeC)
{
  __shared__ float    sAf[64 * 32];   // fast path: raw fp32 A tile [m][k]
  __shared__ float    sBf[64 * 32];   // fast path: raw fp32 B tile
  __shared__ unsigned sA[64][17];     // edge path: packed bf16x2 [m][kpair]
  __shared__ unsigned sB[64][17];     // edge path: packed bf16x2 [n][kpair]

  const int tid  = threadIdx.x;
  const int lane = tid & 31;
  const int wid  = tid >> 5;
  const int wm   = wid & 3;
  const int wn   = wid >> 2;
  const int hi   = lane >> 4;         // lane half selects K sub-block (ISA 7.12.2)
  const int l15  = lane & 15;
  const int m0 = blockIdx.y * 64, n0 = blockIdx.x * 64;
  const int z  = blockIdx.z;
  const long long offA = batchOff(modeA, batA, z);
  const long long offB = batchOff(modeB, batB, z);
  const long long offC = batchOff(modeC, batC, z);

  const int arow = wm * 16 + l15;
  const int bc0  = wn * 32 + l15;
  const int p0 = hi ? 8 : 0;          // A K-pair blocks per lane half
  const int p1 = hi ? 12 : 4;
  const int q0 = hi ? 8 : 0;          // B K-pair block per lane half

  v8f c0 = {0.f,0.f,0.f,0.f,0.f,0.f,0.f,0.f};
  v8f c1 = {0.f,0.f,0.f,0.f,0.f,0.f,0.f,0.f};

  const bool full = (m0 + 64 <= M) && (n0 + 64 <= N) && ((K & 31) == 0);

  if (full) {
    // -------- fast path: async global->LDS staging, branchless --------
    const unsigned baseA = (unsigned)(unsigned long long)(void*)sAf;
    const unsigned baseB = (unsigned)(unsigned long long)(void*)sBf;
    for (int kb = 0; kb < K; kb += 32) {
      // A tile: 64 rows x 32 floats = 512 x 16B chunks, 2 per lane
#pragma unroll
      for (int r = 0; r < 2; ++r) {
        int c = tid + r * 256;
        int row = c >> 3, col = (c & 7) * 4;
        const float* gp = A + offA + (long long)(m0 + row) * lda + kb + col;
        async_g2l_b128(baseA + (unsigned)((row * 32 + col) * 4), gp);
        if (kb + 32 < K) __builtin_prefetch(gp + 32, 0, 1);  // next K tile
      }
      if (transB) {
        // B tile [n][k]: 64 rows x 32 floats, contiguous per row
#pragma unroll
        for (int r = 0; r < 2; ++r) {
          int c = tid + r * 256;
          int row = c >> 3, col = (c & 7) * 4;
          const float* gp = Bm + offB + (long long)(n0 + row) * ldb + kb + col;
          async_g2l_b128(baseB + (unsigned)((row * 32 + col) * 4), gp);
          if (kb + 32 < K) __builtin_prefetch(gp + 32, 0, 1);
        }
      } else {
        // B tile [k][n]: 32 rows x 64 floats, contiguous per row
#pragma unroll
        for (int r = 0; r < 2; ++r) {
          int c = tid + r * 256;
          int row = c >> 4, col = (c & 15) * 4;
          const float* gp = Bm + offB + (long long)(kb + row) * ldb + n0 + col;
          async_g2l_b128(baseB + (unsigned)((row * 64 + col) * 4), gp);
          if (kb + 32 < K) __builtin_prefetch(gp + (long long)32 * ldb, 0, 1);
        }
      }
      wait_async0();
      __syncthreads();

      Frag a, b0, b1;
#pragma unroll
      for (int i = 0; i < 4; ++i) {
        int k0 = (p0 + i) * 2, k1 = (p1 + i) * 2;
        a.u[i]     = pack2(sAf[arow * 32 + k0], sAf[arow * 32 + k0 + 1]);
        a.u[4 + i] = pack2(sAf[arow * 32 + k1], sAf[arow * 32 + k1 + 1]);
      }
      if (transB) {
#pragma unroll
        for (int i = 0; i < 8; ++i) {
          int k = (q0 + i) * 2;
          b0.u[i] = pack2(sBf[bc0 * 32 + k],        sBf[bc0 * 32 + k + 1]);
          b1.u[i] = pack2(sBf[(bc0 + 16) * 32 + k], sBf[(bc0 + 16) * 32 + k + 1]);
        }
      } else {
#pragma unroll
        for (int i = 0; i < 8; ++i) {
          int k = (q0 + i) * 2;
          b0.u[i] = pack2(sBf[k * 64 + bc0],      sBf[(k + 1) * 64 + bc0]);
          b1.u[i] = pack2(sBf[k * 64 + bc0 + 16], sBf[(k + 1) * 64 + bc0 + 16]);
        }
      }
      c0 = __builtin_amdgcn_wmma_f32_16x16x32_bf16(false, a.v, false, b0.v,
                                                   (short)0, c0, false, false);
      c1 = __builtin_amdgcn_wmma_f32_16x16x32_bf16(false, a.v, false, b1.v,
                                                   (short)0, c1, false, false);
      __syncthreads();
    }
  } else {
    // -------- edge path: guarded loads, bf16 conversion on the way in --------
    for (int kb = 0; kb < K; kb += 32) {
      for (int w = tid; w < 1024; w += 256) {
        int r = w >> 4, cp = w & 15;
        int gm = m0 + r, gk = kb + cp * 2;
        float x0 = 0.f, x1 = 0.f;
        if (gm < M) {
          const float* ap = A + offA + (long long)gm * lda + gk;
          if (gk     < K) x0 = ap[0];
          if (gk + 1 < K) x1 = ap[1];
        }
        sA[r][cp] = pack2(x0, x1);
      }
      for (int w = tid; w < 1024; w += 256) {
        int n = w >> 4, cp = w & 15;
        int gn = n0 + n, gk = kb + cp * 2;
        float x0 = 0.f, x1 = 0.f;
        if (gn < N) {
          if (transB) {
            const float* bp = Bm + offB + (long long)gn * ldb + gk;
            if (gk     < K) x0 = bp[0];
            if (gk + 1 < K) x1 = bp[1];
          } else {
            if (gk     < K) x0 = Bm[offB + (long long)gk       * ldb + gn];
            if (gk + 1 < K) x1 = Bm[offB + (long long)(gk + 1) * ldb + gn];
          }
        }
        sB[n][cp] = pack2(x0, x1);
      }
      __syncthreads();

      Frag a, b0, b1;
#pragma unroll
      for (int i = 0; i < 4; ++i) {
        a.u[i]     = sA[arow][p0 + i];
        a.u[4 + i] = sA[arow][p1 + i];
      }
#pragma unroll
      for (int i = 0; i < 8; ++i) {
        b0.u[i] = sB[bc0][q0 + i];
        b1.u[i] = sB[bc0 + 16][q0 + i];
      }
      c0 = __builtin_amdgcn_wmma_f32_16x16x32_bf16(false, a.v, false, b0.v,
                                                   (short)0, c0, false, false);
      c1 = __builtin_amdgcn_wmma_f32_16x16x32_bf16(false, a.v, false, b1.v,
                                                   (short)0, c1, false, false);
      __syncthreads();
    }
  }

  // C/D layout: VGPR i -> M = i + 8*hi, N = lane&15 (ISA 7.12.2)
#pragma unroll
  for (int i = 0; i < 8; ++i) {
    int gm = m0 + wm * 16 + i + (hi ? 8 : 0);
    if (gm >= M) continue;
    int gn0 = n0 + wn * 32 + l15;
    int gn1 = gn0 + 16;
    if (gn0 < N) {
      float v = c0[i] + (bias ? bias[gn0] : 0.f);
      if (relu) v = fmaxf(v, 0.f);
      C[offC + (long long)gm * ldc + gn0] = v;
    }
    if (gn1 < N) {
      float v = c1[i] + (bias ? bias[gn1] : 0.f);
      if (relu) v = fmaxf(v, 0.f);
      C[offC + (long long)gm * ldc + gn1] = v;
    }
  }
}

// ---------------------------------------------------------------------------
// Row softmax with scale + additive mask. rows = B*H*S, cols = 96.
// mode 0: none, 1: src_mask[b, col], 2: causal tgt_mask[t, col]
// ---------------------------------------------------------------------------
__global__ void __launch_bounds__(128)
softmax_rows(float* __restrict__ sc, const float* __restrict__ mask,
             int mode, int cols, float scale)
{
  const int row = blockIdx.x;
  const int b = row / (Hh * Ss);
  const int t = row % Ss;
  const int c = threadIdx.x;
  __shared__ float red[128];
  float val = 0.f, v = -3.0e38f;
  if (c < cols) {
    val = sc[(long long)row * cols + c] * scale;
    if (mode == 1)      val += mask[b * cols + c];
    else if (mode == 2) val += mask[t * cols + c];
    v = val;
  }
  red[c] = v; __syncthreads();
  for (int o = 64; o > 0; o >>= 1) {
    if (c < o) red[c] = fmaxf(red[c], red[c + o]);
    __syncthreads();
  }
  float mx = red[0]; __syncthreads();
  float e = (c < cols) ? __expf(val - mx) : 0.f;
  red[c] = e; __syncthreads();
  for (int o = 64; o > 0; o >>= 1) {
    if (c < o) red[c] += red[c + o];
    __syncthreads();
  }
  float s = red[0];
  if (c < cols) sc[(long long)row * cols + c] = e / s;
}

// ---------------------------------------------------------------------------
// out = LayerNorm(a + res) * g + be   (row = 1024, in-place on res allowed)
// ---------------------------------------------------------------------------
__global__ void __launch_bounds__(256)
add_layernorm(const float* __restrict__ a, const float* __restrict__ res,
              float* __restrict__ out, const float* __restrict__ g,
              const float* __restrict__ be)
{
  const int row = blockIdx.x;
  const int tid = threadIdx.x;
  __shared__ float red[256];
  const long long base = (long long)row * Dm;
  float loc[4]; float s = 0.f;
#pragma unroll
  for (int j = 0; j < 4; ++j) {
    int d = tid + j * 256;
    loc[j] = a[base + d] + res[base + d];
    s += loc[j];
  }
  red[tid] = s; __syncthreads();
  for (int o = 128; o > 0; o >>= 1) { if (tid < o) red[tid] += red[tid + o]; __syncthreads(); }
  float mu = red[0] * (1.f / Dm); __syncthreads();
  float vs = 0.f;
#pragma unroll
  for (int j = 0; j < 4; ++j) { float d2 = loc[j] - mu; vs += d2 * d2; }
  red[tid] = vs; __syncthreads();
  for (int o = 128; o > 0; o >>= 1) { if (tid < o) red[tid] += red[tid + o]; __syncthreads(); }
  float rstd = rsqrtf(red[0] * (1.f / Dm) + 1e-5f);
#pragma unroll
  for (int j = 0; j < 4; ++j) {
    int d = tid + j * 256;
    out[base + d] = (loc[j] - mu) * rstd * g[d] + be[d];
  }
}

// ---------------------------------------------------------------------------
// Embedding gather * sqrt(D) + sinusoidal positional encoding (computed here,
// since _POS is a closure constant, not an input).
// ---------------------------------------------------------------------------
__global__ void __launch_bounds__(256)
embed_pos(const int* __restrict__ tok, const float* __restrict__ table,
          float* __restrict__ out)
{
  const int row = blockIdx.x;          // B*S rows
  const int s   = row % Ss;
  const int tid = threadIdx.x;
  const int id  = tok[row];
  const float pos = (float)s;
  const long long tb = (long long)id * Dm;
  const long long ob = (long long)row * Dm;
#pragma unroll
  for (int j = 0; j < 4; ++j) {
    int d  = tid + j * 256;
    int de = d & ~1;
    float div = __expf(-(float)de * (9.2103403719761840f / (float)Dm));
    float pe  = (d & 1) ? __cosf(pos * div) : __sinf(pos * div);
    out[ob + d] = table[tb + d] * 32.0f + pe;   // sqrt(1024) = 32
  }
}

// ---------------------------------------------------------------------------
extern "C" void kernel_launch(void* const* d_in, const int* in_sizes, int n_in,
                              void* d_out, int out_size, void* d_ws, size_t ws_size,
                              hipStream_t stream)
{
  (void)in_sizes; (void)n_in; (void)out_size; (void)ws_size;

  // top-level inputs (insertion order)
  const int*   src      = (const int*)  d_in[0];
  const int*   tgt      = (const int*)  d_in[1];
  const float* src_mask = (const float*)d_in[2];   // [B,1,1,S]
  const float* tgt_mask = (const float*)d_in[3];   // [1,1,T,T]
  const float* src_emb  = (const float*)d_in[4];
  const float* tgt_emb  = (const float*)d_in[5];
  const float* out_w    = (const float*)d_in[6];
  const float* out_b    = (const float*)d_in[7];

  // enc_params leaves, alphabetical (JAX dict flatten order)
  const float* e_b1  = (const float*)d_in[ 8];
  const float* e_b2  = (const float*)d_in[ 9];
  const float* e_be1 = (const float*)d_in[10];
  const float* e_be2 = (const float*)d_in[11];
  const float* e_bk  = (const float*)d_in[12];
  const float* e_bo  = (const float*)d_in[13];
  const float* e_bq  = (const float*)d_in[14];
  const float* e_bv  = (const float*)d_in[15];
  const float* e_g1  = (const float*)d_in[16];
  const float* e_g2  = (const float*)d_in[17];
  const float* e_w1  = (const float*)d_in[18];
  const float* e_w2  = (const float*)d_in[19];
  const float* e_wk  = (const float*)d_in[20];
  const float* e_wo  = (const float*)d_in[21];
  const float* e_wq  = (const float*)d_in[22];
  const float* e_wv  = (const float*)d_in[23];

  // dec_params leaves, alphabetical
  const float* dd_b1  = (const float*)d_in[24];
  const float* dd_b2  = (const float*)d_in[25];
  const float* dd_be1 = (const float*)d_in[26];
  const float* dd_be2 = (const float*)d_in[27];
  const float* dd_be3 = (const float*)d_in[28];
  const float* c_bk   = (const float*)d_in[29];
  const float* c_bo   = (const float*)d_in[30];
  const float* c_bq   = (const float*)d_in[31];
  const float* c_bv   = (const float*)d_in[32];
  const float* c_wk   = (const float*)d_in[33];
  const float* c_wo   = (const float*)d_in[34];
  const float* c_wq   = (const float*)d_in[35];
  const float* c_wv   = (const float*)d_in[36];
  const float* dd_g1  = (const float*)d_in[37];
  const float* dd_g2  = (const float*)d_in[38];
  const float* dd_g3  = (const float*)d_in[39];
  const float* s_bk   = (const float*)d_in[40];
  const float* s_bo   = (const float*)d_in[41];
  const float* s_bq   = (const float*)d_in[42];
  const float* s_bv   = (const float*)d_in[43];
  const float* s_wk   = (const float*)d_in[44];
  const float* s_wo   = (const float*)d_in[45];
  const float* s_wq   = (const float*)d_in[46];
  const float* s_wv   = (const float*)d_in[47];
  const float* dd_w1  = (const float*)d_in[48];
  const float* dd_w2  = (const float*)d_in[49];

  // workspace layout (floats)
  float* ws = (float*)d_ws;
  const long long AD = (long long)MT * Dm;          // 3.1M floats
  float* X  = ws;                                   // encoder stream
  float* Y  = X  + AD;                              // decoder stream
  float* Q  = Y  + AD;                              // q / generic temp
  float* Kb = Q  + AD;                              // k
  float* Vb = Kb + AD;                              // v
  float* Cx = Vb + AD;                              // attention context
  float* Hb = Cx + AD;                              // FFN hidden 3072x4096
  float* SC = Hb + (long long)MT * DFFm;            // scores 512x96x96

  auto gemm = [&](const float* A, const float* Bm, float* C, const float* bias,
                  int M, int N, int K, int lda, int ldb, int ldc,
                  int transB, int relu, int batches,
                  long long batA, long long batB, long long batC,
                  int modeA, int modeB, int modeC) {
    dim3 grid((unsigned)((N + 63) / 64), (unsigned)((M + 63) / 64), (unsigned)batches);
    gemm_bf16_wmma<<<grid, dim3(256), 0, stream>>>(
        A, Bm, C, bias, M, N, K, lda, ldb, ldc, transB, relu,
        batA, batB, batC, modeA, modeB, modeC);
  };

  auto mha = [&](const float* xq, const float* xkv,
                 const float* wq, const float* bq, const float* wk, const float* bk,
                 const float* wv, const float* bv, const float* wo, const float* bo,
                 const float* mask, int maskmode) {
    gemm(xq,  wq, Q,  bq, MT, Dm, Dm, Dm, Dm, Dm, 0, 0, 1, 0, 0, 0, 0, 0, 0);
    gemm(xkv, wk, Kb, bk, MT, Dm, Dm, Dm, Dm, Dm, 0, 0, 1, 0, 0, 0, 0, 0, 0);
    gemm(xkv, wv, Vb, bv, MT, Dm, Dm, Dm, Dm, Dm, 0, 0, 1, 0, 0, 0, 0, 0, 0);
    // scores[b,h] = Q_head(96x64) x K_head^T  -> SC (batch-linear 96x96)
    gemm(Q, Kb, SC, nullptr, Ss, Ss, DHh, Dm, Dm, Ss, 1, 0, Bb * Hh,
         0, 0, (long long)Ss * Ss, 1, 1, 0);
    softmax_rows<<<Bb * Hh * Ss, 128, 0, stream>>>(SC, mask, maskmode, Ss, 0.125f);
    // ctx[b,h] = attn(96x96) x V_head(96x64), written into head view of Cx
    gemm(SC, Vb, Cx, nullptr, Ss, DHh, Ss, Ss, Dm, Dm, 0, 0, Bb * Hh,
         (long long)Ss * Ss, 0, 0, 0, 1, 1);
    // output projection -> Q (q buffer is dead by now)
    gemm(Cx, wo, Q, bo, MT, Dm, Dm, Dm, Dm, Dm, 0, 0, 1, 0, 0, 0, 0, 0, 0);
  };

  auto ffn = [&](float* xio, const float* w1, const float* b1,
                 const float* w2, const float* b2,
                 const float* g, const float* be) {
    gemm(xio, w1, Hb, b1, MT, DFFm, Dm, Dm, DFFm, DFFm, 0, 1, 1, 0, 0, 0, 0, 0, 0);
    gemm(Hb, w2, Q, b2, MT, Dm, DFFm, DFFm, Dm, Dm, 0, 0, 1, 0, 0, 0, 0, 0, 0);
    add_layernorm<<<MT, 256, 0, stream>>>(Q, xio, xio, g, be);
  };

  const long long WD = (long long)Dm * Dm;
  const long long FD = (long long)Dm * DFFm;

  // ---------------- encoder ----------------
  embed_pos<<<MT, 256, 0, stream>>>(src, src_emb, X);
  for (int i = 0; i < NLAY; ++i) {
    mha(X, X,
        e_wq + i * WD, e_bq + i * Dm, e_wk + i * WD, e_bk + i * Dm,
        e_wv + i * WD, e_bv + i * Dm, e_wo + i * WD, e_bo + i * Dm,
        src_mask, 1);
    add_layernorm<<<MT, 256, 0, stream>>>(Q, X, X, e_g1 + i * Dm, e_be1 + i * Dm);
    ffn(X, e_w1 + i * FD, e_b1 + (long long)i * DFFm,
        e_w2 + i * FD, e_b2 + i * Dm, e_g2 + i * Dm, e_be2 + i * Dm);
  }

  // ---------------- decoder ----------------
  embed_pos<<<MT, 256, 0, stream>>>(tgt, tgt_emb, Y);
  for (int i = 0; i < NLAY; ++i) {
    // causal self-attention
    mha(Y, Y,
        s_wq + i * WD, s_bq + i * Dm, s_wk + i * WD, s_bk + i * Dm,
        s_wv + i * WD, s_bv + i * Dm, s_wo + i * WD, s_bo + i * Dm,
        tgt_mask, 2);
    add_layernorm<<<MT, 256, 0, stream>>>(Q, Y, Y, dd_g1 + i * Dm, dd_be1 + i * Dm);
    // cross-attention (no mask)
    mha(Y, X,
        c_wq + i * WD, c_bq + i * Dm, c_wk + i * WD, c_bk + i * Dm,
        c_wv + i * WD, c_bv + i * Dm, c_wo + i * WD, c_bo + i * Dm,
        nullptr, 0);
    add_layernorm<<<MT, 256, 0, stream>>>(Q, Y, Y, dd_g2 + i * Dm, dd_be2 + i * Dm);
    ffn(Y, dd_w1 + i * FD, dd_b1 + (long long)i * DFFm,
        dd_w2 + i * FD, dd_b2 + i * Dm, dd_g3 + i * Dm, dd_be3 + i * Dm);
  }

  // ---------------- logits ----------------
  gemm(Y, out_w, (float*)d_out, out_b, MT, VT, Dm, Dm, VT, VT,
       0, 0, 1, 0, 0, 0, 0, 0, 0);
}
